// SpinEmbedding_44057774522739
// MI455X (gfx1250) — matrix operand order repaired and checked by reading, not compiled, not run on hardware
//
#include <hip/hip_runtime.h>
#include <hip/hip_bf16.h>
#include <math.h>

#define D 128

typedef __attribute__((ext_vector_type(16))) __bf16 v16bf;
typedef __attribute__((ext_vector_type(8)))  float  v8f;

// Hardware-native transcendentals: v_exp_f32 / v_rcp_f32 (~1 ulp), far below
// the bf16 WMMA quantization already in the h2 term.
__device__ __forceinline__ float fast_rcp(float x) { return __builtin_amdgcn_rcpf(x); }
__device__ __forceinline__ float siluf(float x) {
    return x * fast_rcp(1.0f + __expf(-x));
}
__device__ __forceinline__ float softplusf(float x) {
    return (x > 20.0f) ? x : __logf(1.0f + __expf(x));
}
__device__ __forceinline__ unsigned short f2bf(float f) {
    union { __bf16 b; unsigned short s; } t;
    t.b = (__bf16)f;
    return t.s;
}

// ---------------------------------------------------------------------------
// prep: u = Wq^T @ Wk (D), c = bq . Wk, w1v = W1 @ Wv (D), W2 -> bf16
// One block, 128 threads. All matrices are tiny (64 KB max), L2-resident.
// ---------------------------------------------------------------------------
__global__ void spin_prep(const float* __restrict__ Wq, const float* __restrict__ bq,
                          const float* __restrict__ Wk, const float* __restrict__ Wv,
                          const float* __restrict__ W1, const float* __restrict__ W2,
                          float* __restrict__ u, float* __restrict__ cptr,
                          float* __restrict__ w1v, unsigned short* __restrict__ W2bf) {
    __shared__ float red[D];
    const int t = threadIdx.x;   // 0..127

    // u[t] = sum_d Wq[d, t] * Wk[d]   (coalesced across threads for each d)
    float acc = 0.0f;
    for (int d = 0; d < D; ++d) acc += Wq[d * D + t] * Wk[d];
    u[t] = acc;

    // c = sum_d bq[d] * Wk[d]
    red[t] = bq[t] * Wk[t];
    __syncthreads();
    for (int s = 64; s > 0; s >>= 1) {
        if (t < s) red[t] += red[t + s];
        __syncthreads();
    }
    if (t == 0) cptr[0] = red[0];

    // w1v[t] = sum_e W1[t, e] * Wv[e]
    float acc2 = 0.0f;
    for (int e = 0; e < D; ++e) acc2 += W1[t * D + e] * Wv[e];
    w1v[t] = acc2;

    // W2 -> bf16 row-major (B operand source)
    for (int i = 0; i < D; ++i) {
        int idx = t * D + i;
        W2bf[idx] = f2bf(W2[idx]);
    }
}

__global__ void spin_zero(float* __restrict__ asum, int G) {
    int i = blockIdx.x * blockDim.x + threadIdx.x;
    if (i < G) asum[i] = 0.0f;
}

// ---------------------------------------------------------------------------
// pass1: attn_n = softplus(scale * s_g * (node_scalar[n] . u + c))
//        attn_sum[g] += attn_n   (global f32 atomic, resolved at L2)
// 4 lanes cooperate per node (32 floats each, float4 loads).
// ---------------------------------------------------------------------------
__global__ void spin_pass1(const float* __restrict__ ns, const int* __restrict__ batch,
                           const float* __restrict__ spin, const float* __restrict__ u,
                           const float* __restrict__ cptr, float* __restrict__ attn,
                           float* __restrict__ asum, int N) {
    __shared__ float us[D];
    __shared__ float csh;
    const int t = threadIdx.x;
    if (t < D) us[t] = u[t];
    if (t == 0) csh = cptr[0];
    __syncthreads();

    const int gid  = blockIdx.x * blockDim.x + t;
    const int node = gid >> 2;
    const int sub  = gid & 3;
    if (node >= N) return;

    const float4* p  = (const float4*)(ns + (size_t)node * D + sub * 32);
    const float4* uq = (const float4*)(us + sub * 32);
    float s = 0.0f;
#pragma unroll
    for (int i = 0; i < 8; ++i) {
        float4 a = p[i], b = uq[i];
        s += a.x * b.x + a.y * b.y + a.z * b.z + a.w * b.w;
    }
    s += __shfl_xor(s, 1, 32);
    s += __shfl_xor(s, 2, 32);

    if (sub == 0) {
        const int g   = batch[node];
        const float sp = spin[g];
        const float sg = sp / fmaxf(sp, 1.0f);
        const float dot = sg * (s + csh) * 0.08838834764831845f;  // 1/sqrt(128)
        const float at = softplusf(dot);
        attn[node] = at;
        atomicAdd(&asum[g], at);
    }
}

// ---------------------------------------------------------------------------
// pass2: out[n,:] = ns[n,:] + a_n*Wv + silu( h1(a_n) @ W2^T + b2 )
//   h1[n,k] = silu(a_n*w1v[k] + b1[k])  -- built directly into WMMA A frags.
// 256 threads = 8 waves; wave w owns a 16-node tile; 32 wmma per wave
// (8 column tiles x 4 K-chunks), f32 accumulation.
// Fast-path epilogue (all blocks except the tail): transpose the 16x16 f32
// result tile through padded LDS, then float4 ns loads + float4 out stores.
// The fullBlock branch is hoisted OUTSIDE the ct loop (block-uniform).
// ---------------------------------------------------------------------------
#define TSTRIDE 20   // 16 + 4 pad: half-waves hit disjoint bank ranges

__global__ void spin_pass2(const float* __restrict__ ns, const int* __restrict__ batch,
                           const float* __restrict__ spin, const float* __restrict__ Wv,
                           const float* __restrict__ b1, const float* __restrict__ b2,
                           const float* __restrict__ attn, const float* __restrict__ asum,
                           const float* __restrict__ w1v, const unsigned int* __restrict__ W2p,
                           float* __restrict__ out, int N) {
    __shared__ unsigned int w2s[D * D / 2];  // row n: 64 dwords = 128 bf16 (k pairs)
    __shared__ float w1vs[D], b1s[D], b2s[D], wvs[D], as[128];
    __shared__ __align__(16) float tbuf[8 * 16 * TSTRIDE];  // per-wave 16x16 staging

    const int t = threadIdx.x;
    for (int i = t; i < D * D / 2; i += 256) w2s[i] = W2p[i];
    if (t < D) { w1vs[t] = w1v[t]; b1s[t] = b1[t]; b2s[t] = b2[t]; wvs[t] = Wv[t]; }

    const int node0 = blockIdx.x * 128;
    if (t < 128) {
        const int node = node0 + t;
        float av = 0.0f;
        if (node < N) {
            const int g = batch[node];
            av = attn[node] * spin[g] / asum[g];
        }
        as[t] = av;
    }
    __syncthreads();

    const int wave = t >> 5;       // 0..7 -> 16-node tile
    const int lane = t & 31;
    const int tileBase = wave * 16;
    const int mrow = lane & 15;
    const float am = as[tileBase + mrow];
    const int khalf = (lane < 16) ? 0 : 8;     // unused; A frag also uses kb16
    const int kb16  = (lane < 16) ? 0 : 16;    // B-frag K offset per half-wave
    const int ncol  = lane & 15;
    const int mofs  = (lane < 16) ? 0 : 8;     // C/D row offset per half-wave

    // Build A fragments: 16-bit A 16x32, lane = M%16,
    // VGPR v: K = 2v (+16 for hi lanes), matching the B fragment K order.
    v16bf afrag[4];
#pragma unroll
    for (int kc = 0; kc < 4; ++kc) {
#pragma unroll
        for (int v = 0; v < 8; ++v) {
            const int kb = 2 * v + kb16;
            const int k0 = kc * 32 + kb;
            afrag[kc][2 * v]     = (__bf16)siluf(am * w1vs[k0]     + b1s[k0]);
            afrag[kc][2 * v + 1] = (__bf16)siluf(am * w1vs[k0 + 1] + b1s[k0 + 1]);
        }
    }

    // Hoisted per-row a values for the epilogue (C/D rows this lane owns).
    float amv[8];
#pragma unroll
    for (int r = 0; r < 8; ++r) amv[r] = as[tileBase + r + mofs];

    float* tb = tbuf + wave * 16 * TSTRIDE;
    const int trow  = lane >> 1;       // transpose read: row 0..15
    const int thalf = lane & 1;        // col group 0/1 (8 floats each)

    union BFU { v16bf v; unsigned int u[8]; };

    if (node0 + 128 <= N) {
        // ---- fast path: full 128-node block, no predication anywhere ----
#pragma unroll 2
        for (int ct = 0; ct < 8; ++ct) {
            v8f acc = {};
            const int n = ct * 16 + ncol;            // output column (= W2 row)
            const int rowbase = n * (D / 2);
#pragma unroll
            for (int kc = 0; kc < 4; ++kc) {
                // B 32x16 bf16: lanes 0-15 K=0..15, lanes 16-31 K=16..31.
                BFU bf;
                const int pbase = rowbase + (kc * 32 + kb16) / 2;
#pragma unroll
                for (int v = 0; v < 8; ++v) bf.u[v] = w2s[pbase + v];
                acc = __builtin_amdgcn_wmma_f32_16x16x32_bf16(
                    false, afrag[kc], false, bf.v, (short)0, acc, false, false);
            }

            const float b2d = b2s[n];
            const float wvd = wvs[n];
            // spin_embed -> padded LDS (conflict-free: half-waves hit bank
            // ranges B..B+15 and B+32..B+47).
#pragma unroll
            for (int r = 0; r < 8; ++r) {
                tb[(r + mofs) * TSTRIDE + ncol] = amv[r] * wvd + siluf(acc[r] + b2d);
            }
            // Same-wave readback (in-order LDS): each lane owns 8 contiguous
            // floats of one row -> float4 ns load + float4 out store.
            const int nodeR = node0 + tileBase + trow;
            const size_t base = (size_t)nodeR * D + ct * 16 + thalf * 8;
            const float4* nsr  = (const float4*)(ns + base);
            float4*       orow = (float4*)(out + base);
            const float4* trd  = (const float4*)(tb + trow * TSTRIDE + thalf * 8);
#pragma unroll
            for (int j = 0; j < 2; ++j) {
                float4 nv = nsr[j];
                float4 ev = trd[j];
                float4 o;
                o.x = nv.x + ev.x; o.y = nv.y + ev.y;
                o.z = nv.z + ev.z; o.w = nv.w + ev.w;
                orow[j] = o;
            }
        }
    } else {
        // ---- tail block only: guarded scalar epilogue ----
#pragma unroll 2
        for (int ct = 0; ct < 8; ++ct) {
            v8f acc = {};
            const int n = ct * 16 + ncol;
            const int rowbase = n * (D / 2);
#pragma unroll
            for (int kc = 0; kc < 4; ++kc) {
                BFU bf;
                const int pbase = rowbase + (kc * 32 + kb16) / 2;
#pragma unroll
                for (int v = 0; v < 8; ++v) bf.u[v] = w2s[pbase + v];
                acc = __builtin_amdgcn_wmma_f32_16x16x32_bf16(
                    false, afrag[kc], false, bf.v, (short)0, acc, false, false);
            }
            const float b2d = b2s[n];
            const float wvd = wvs[n];
#pragma unroll
            for (int r = 0; r < 8; ++r) {
                const int node = node0 + tileBase + r + mofs;
                if (node < N) {
                    const float val = amv[r] * wvd + siluf(acc[r] + b2d);
                    const size_t idx = (size_t)node * D + n;
                    out[idx] = ns[idx] + val;
                }
            }
        }
    }
}

// ---------------------------------------------------------------------------
extern "C" void kernel_launch(void* const* d_in, const int* in_sizes, int n_in,
                              void* d_out, int out_size, void* d_ws, size_t ws_size,
                              hipStream_t stream) {
    const float* node_scalar = (const float*)d_in[0];
    const int*   batch       = (const int*)d_in[1];
    const float* spin        = (const float*)d_in[2];
    const float* Wq          = (const float*)d_in[3];
    const float* bq          = (const float*)d_in[4];
    const float* Wk          = (const float*)d_in[5];
    const float* Wv          = (const float*)d_in[6];
    const float* W1          = (const float*)d_in[7];
    const float* b1          = (const float*)d_in[8];
    const float* W2          = (const float*)d_in[9];
    const float* b2          = (const float*)d_in[10];
    float* out = (float*)d_out;

    const int N = in_sizes[0] / D;   // 200000
    const int G = in_sizes[2];       // 2000

    // workspace layout (floats): attn[N] | asum[G] | u[D] | c | w1v[D] | W2bf(ushort[D*D])
    float* ws   = (float*)d_ws;
    float* attn = ws;
    float* asum = ws + N;
    float* u    = asum + G;
    float* cptr = u + D;
    float* w1v  = cptr + 1;
    unsigned short* W2bf = (unsigned short*)(w1v + D);

    hipLaunchKernelGGL(spin_prep, dim3(1), dim3(D), 0, stream,
                       Wq, bq, Wk, Wv, W1, W2, u, cptr, w1v, W2bf);

    hipLaunchKernelGGL(spin_zero, dim3((G + 255) / 256), dim3(256), 0, stream, asum, G);

    const int p1_threads = 4 * N;
    hipLaunchKernelGGL(spin_pass1, dim3((p1_threads + 255) / 256), dim3(256), 0, stream,
                       node_scalar, batch, spin, u, cptr, attn, asum, N);

    hipLaunchKernelGGL(spin_pass2, dim3((N + 127) / 128), dim3(256), 0, stream,
                       node_scalar, batch, spin, Wv, b1, b2,
                       attn, asum, w1v, (const unsigned int*)W2bf, out, N);
}
